// EnhancedDiacritizer_38732015075738
// MI455X (gfx1250) — matrix operand order repaired and checked by reading, not compiled, not run on hardware
//
#include <hip/hip_runtime.h>
#include <hip/hip_bf16.h>

typedef _Float16 h16;
typedef __attribute__((ext_vector_type(16))) _Float16 v16h;
typedef __attribute__((ext_vector_type(8)))  float    v8f;

union FragH { v16h v; unsigned int u[8]; h16 h[16]; };
union FragF { v8f  v; float f[8]; };

static __device__ __forceinline__ float sigmf(float x) {
    return 1.0f / (1.0f + __expf(-x));
}

// ---------------------------------------------------------------------------
// Utility kernels
// ---------------------------------------------------------------------------
__global__ void k_zero(float* p, int n) {
    int i = blockIdx.x * 256 + threadIdx.x;
    if (i < n) p[i] = 0.0f;
}

__global__ void k_cvt(const float* __restrict__ s, h16* __restrict__ d, int n) {
    int i = blockIdx.x * 256 + threadIdx.x;
    if (i < n) d[i] = (h16)s[i];
}

// src [rows, ksrc] f32 -> dst [rows, kdst] f16, zero padded K
__global__ void k_cvt_pad(const float* __restrict__ s, h16* __restrict__ d,
                          int rows, int ksrc, int kdst) {
    int i = blockIdx.x * 256 + threadIdx.x;
    int n = rows * kdst;
    if (i >= n) return;
    int r = i / kdst, k = i - r * kdst;
    d[i] = (h16)((k < ksrc) ? s[(size_t)r * ksrc + k] : 0.0f);
}

// src [K,N] f32 -> dst[(rowoff+n)*ldd + k] f16  (transpose to N-major)
__global__ void k_cvt_T(const float* __restrict__ s, h16* __restrict__ d,
                        int K, int N, int rowoff, int ldd) {
    int i = blockIdx.x * 256 + threadIdx.x;
    if (i >= K * N) return;
    int n = i / K, k = i - n * K;
    d[(size_t)(rowoff + n) * ldd + k] = (h16)s[(size_t)k * N + n];
}

// ---------------------------------------------------------------------------
// Token -> word scatter-mean pooling (sums + counts via atomics)
// ---------------------------------------------------------------------------
__global__ void k_pool(const int* __restrict__ word_ids,
                       const float* __restrict__ tok,
                       float* __restrict__ sums, float* __restrict__ cnts) {
    int bs = blockIdx.x;                // b*384 + s
    int b = bs / 384;
    int wid = word_ids[bs];
    if (wid < 0 || wid >= 64) return;
    const float* t = tok + (size_t)bs * 768;
    float* d = sums + (size_t)(b * 64 + wid) * 768;
    for (int j = threadIdx.x; j < 768; j += 256) atomicAdd(&d[j], t[j]);
    if (threadIdx.x == 0) atomicAdd(&cnts[b * 64 + wid], 1.0f);
}

// ---------------------------------------------------------------------------
// Build LSTM layer-0 input X [8192, 1120] f16:
//  [0:300) char emb | [300:348) feat-proj MLP | [348:1116) bert ctx | pad
// ---------------------------------------------------------------------------
__global__ void k_prep(const int* __restrict__ char_ids, const int* __restrict__ char2word,
                       const float* __restrict__ feats, const float* __restrict__ emb_table,
                       const float* __restrict__ fp_W1, const float* __restrict__ fp_b1,
                       const float* __restrict__ fp_ln_g, const float* __restrict__ fp_ln_b,
                       const float* __restrict__ fp_W2, const float* __restrict__ fp_b2,
                       const float* __restrict__ wsums, const float* __restrict__ wcnts,
                       h16* __restrict__ X) {
    const int m = blockIdx.x;          // b*512 + t
    const int b = m >> 9;
    const int tid = threadIdx.x;       // 128
    __shared__ float s1[48], s2[48];

    int cid = char_ids[m];
    for (int j = tid; j < 300; j += 128)
        X[(size_t)m * 1120 + j] = (h16)emb_table[cid * 300 + j];

    if (tid < 48) {
        float s = fp_b1[tid];
        for (int k = 0; k < 24; ++k) s += feats[m * 24 + k] * fp_W1[k * 48 + tid];
        s1[tid] = s;
    }
    __syncthreads();
    if (tid < 48) {
        float mean = 0.f, var = 0.f;
        for (int k = 0; k < 48; ++k) mean += s1[k];
        mean *= (1.f / 48.f);
        for (int k = 0; k < 48; ++k) { float dd = s1[k] - mean; var += dd * dd; }
        var *= (1.f / 48.f);
        float hn = (s1[tid] - mean) * rsqrtf(var + 1e-5f) * fp_ln_g[tid] + fp_ln_b[tid];
        s2[tid] = fmaxf(hn, 0.f);
    }
    __syncthreads();
    if (tid < 48) {
        float s = fp_b2[tid];
        for (int k = 0; k < 48; ++k) s += s2[k] * fp_W2[k * 48 + tid];
        X[(size_t)m * 1120 + 300 + tid] = (h16)fmaxf(s, 0.f);
    }

    int wid = char2word[m];
    if (wid >= 0 && wid < 64) {
        float inv = 1.f / fmaxf(wcnts[b * 64 + wid], 1.f);
        const float* ws = wsums + (size_t)(b * 64 + wid) * 768;
        for (int j = tid; j < 768; j += 128)
            X[(size_t)m * 1120 + 348 + j] = (h16)(ws[j] * inv);
    } else {
        for (int j = tid; j < 768; j += 128)
            X[(size_t)m * 1120 + 348 + j] = (h16)0.f;
    }
    if (tid < 4) X[(size_t)m * 1120 + 1116 + tid] = (h16)0.f;
}

// ---------------------------------------------------------------------------
// WMMA GEMM: C[M,N] (f32) = A[M,K] (f16, row major) x B[N,K]^T (f16, N-major)
// block = 128 threads = 4 waves; each wave computes 4 consecutive 16x16 N-tiles
// (A fragment reused across the 4 WMMAs). grid = (N/256, M/16).
// ---------------------------------------------------------------------------
__global__ void __launch_bounds__(128)
k_gemm_f16(const h16* __restrict__ A, const h16* __restrict__ B,
           float* __restrict__ C, int N, int K, int lda, int ldb) {
    const int lane = threadIdx.x & 31;
    const int wave = threadIdx.x >> 5;
    const int tile_n0 = (blockIdx.x * 4 + wave) * 4;   // first of 4 N-tiles
    const int tile_m = blockIdx.y;
    const int klo = (lane < 16) ? 0 : 16;
    const int khi = (lane < 16) ? 8 : 24;
    const int bk  = (lane < 16) ? 0 : 16;
    const h16* Arow = A + (size_t)(tile_m * 16 + (lane & 15)) * lda;
    const h16* Brow[4];
#pragma unroll
    for (int j = 0; j < 4; ++j)
        Brow[j] = B + (size_t)((tile_n0 + j) * 16 + (lane & 15)) * ldb;

    FragF acc[4];
#pragma unroll
    for (int j = 0; j < 4; ++j)
#pragma unroll
        for (int i = 0; i < 8; ++i) acc[j].f[i] = 0.f;

    for (int k0 = 0; k0 < K; k0 += 32) {
        FragH a;
        const unsigned int* pl = (const unsigned int*)(Arow + k0 + klo);
        const unsigned int* ph = (const unsigned int*)(Arow + k0 + khi);
#pragma unroll
        for (int i = 0; i < 4; ++i) { a.u[i] = pl[i]; a.u[4 + i] = ph[i]; }
#pragma unroll
        for (int j = 0; j < 4; ++j) {
            FragH bfr;
            const unsigned int* pb = (const unsigned int*)(Brow[j] + k0 + bk);
#pragma unroll
            for (int i = 0; i < 8; ++i) bfr.u[i] = pb[i];
            acc[j].v = __builtin_amdgcn_wmma_f32_16x16x32_f16(
                false, a.v, false, bfr.v, (short)0, acc[j].v, false, false);
        }
    }
#pragma unroll
    for (int j = 0; j < 4; ++j) {
        const int col = (tile_n0 + j) * 16 + (lane & 15);
#pragma unroll
        for (int r = 0; r < 8; ++r) {
            int row = tile_m * 16 + ((lane < 16) ? r : 8 + r);
            C[(size_t)row * N + col] = acc[j].f[r];
        }
    }
}

// ---------------------------------------------------------------------------
// Bidirectional LSTM recurrence, one layer. grid=2 (dir), block=1024 (32 waves).
// Zpre [8192, 2048] f32 holds x@Wih^T (col offset dir*1024).
// Whh f16 [2048, 256] N-major.  bias f32 [2][2][1024] (b_ih, b_hh per dir).
// yout f16 [8192, 512] = concat(fwd, bwd).
// LDS: z (16x1024 f32) + h (16x256 f16). c-state in registers.
// Persistent-RNN: each wave pins its 2 N-tiles of Whh fragments in VGPRs
// (2 tiles x 8 k-steps x 8 VGPRs = 128 VGPRs) for all 512 timesteps.
// ---------------------------------------------------------------------------
__global__ void __launch_bounds__(1024)
k_lstm_recur(const float* __restrict__ Zpre, const h16* __restrict__ Whh,
             const float* __restrict__ bias, h16* __restrict__ yout) {
    extern __shared__ char smem[];
    float* zbuf = (float*)smem;                    // 16*1024 f32
    h16*   hbuf = (h16*)(smem + 16 * 1024 * 4);    // 16*256 f16

    const int dir  = blockIdx.x;
    const int tid  = threadIdx.x;
    const int lane = tid & 31;
    const int wave = tid >> 5;
    const int mrow = lane & 15;
    const int klo  = (lane < 16) ? 0 : 16;
    const int khi  = (lane < 16) ? 8 : 24;
    const int bk   = (lane < 16) ? 0 : 16;

    const h16* w0 = Whh + (size_t)(dir * 1024 + (wave * 2 + 0) * 16 + (lane & 15)) * 256;
    const h16* w1 = Whh + (size_t)(dir * 1024 + (wave * 2 + 1) * 16 + (lane & 15)) * 256;
    const float* bi = bias + dir * 2048;

    // Pin Whh fragments in registers for the whole sequence.
    FragH wb0[8], wb1[8];
#pragma unroll
    for (int kk = 0; kk < 8; ++kk) {
        const unsigned int* p0 = (const unsigned int*)(w0 + kk * 32 + bk);
        const unsigned int* p1 = (const unsigned int*)(w1 + kk * 32 + bk);
#pragma unroll
        for (int i = 0; i < 8; ++i) { wb0[kk].u[i] = p0[i]; wb1[kk].u[i] = p1[i]; }
    }

    for (int i = tid; i < 16 * 256; i += 1024) hbuf[i] = (h16)0.f;
    float cst[4] = {0.f, 0.f, 0.f, 0.f};
    __syncthreads();

    for (int t = 0; t < 512; ++t) {
        const int ta = dir ? (511 - t) : t;

        FragF acc0, acc1;
#pragma unroll
        for (int i = 0; i < 8; ++i) { acc0.f[i] = 0.f; acc1.f[i] = 0.f; }

#pragma unroll
        for (int kk = 0; kk < 8; ++kk) {
            FragH a;
            const unsigned int* pl = (const unsigned int*)(hbuf + mrow * 256 + kk * 32 + klo);
            const unsigned int* ph = (const unsigned int*)(hbuf + mrow * 256 + kk * 32 + khi);
#pragma unroll
            for (int i = 0; i < 4; ++i) { a.u[i] = pl[i]; a.u[4 + i] = ph[i]; }
            acc0.v = __builtin_amdgcn_wmma_f32_16x16x32_f16(
                false, a.v, false, wb0[kk].v, (short)0, acc0.v, false, false);
            acc1.v = __builtin_amdgcn_wmma_f32_16x16x32_f16(
                false, a.v, false, wb1[kk].v, (short)0, acc1.v, false, false);
        }
#pragma unroll
        for (int r = 0; r < 8; ++r) {
            int row = (lane < 16) ? r : 8 + r;
            zbuf[row * 1024 + (wave * 2 + 0) * 16 + (lane & 15)] = acc0.f[r];
            zbuf[row * 1024 + (wave * 2 + 1) * 16 + (lane & 15)] = acc1.f[r];
        }
        __syncthreads();

#pragma unroll
        for (int i = 0; i < 4; ++i) {
            int flat = i * 1024 + tid;
            int b = flat >> 8, j = flat & 255;
            size_t zb = (size_t)(b * 512 + ta) * 2048 + (size_t)dir * 1024;
            float zi = zbuf[b * 1024 + j]       + Zpre[zb + j]       + bi[j]         + bi[1024 + j];
            float zf = zbuf[b * 1024 + 256 + j] + Zpre[zb + 256 + j] + bi[256 + j]   + bi[1280 + j];
            float zg = zbuf[b * 1024 + 512 + j] + Zpre[zb + 512 + j] + bi[512 + j]   + bi[1536 + j];
            float zo = zbuf[b * 1024 + 768 + j] + Zpre[zb + 768 + j] + bi[768 + j]   + bi[1792 + j];
            float c  = sigmf(zf) * cst[i] + sigmf(zi) * tanhf(zg);
            cst[i] = c;
            float h = sigmf(zo) * tanhf(c);
            hbuf[b * 256 + j] = (h16)h;
            yout[(size_t)(b * 512 + ta) * 512 + dir * 256 + j] = (h16)h;
        }
        __syncthreads();
    }
}

// ---------------------------------------------------------------------------
// LayerNorm over last dim 512 (f16 in -> f16 out). grid=8192, block=256.
// ---------------------------------------------------------------------------
__global__ void k_layernorm512(const h16* __restrict__ y, const float* __restrict__ g,
                               const float* __restrict__ bta, h16* __restrict__ out) {
    int m = blockIdx.x, tid = threadIdx.x;
    __shared__ float red[256];
    float v0 = (float)y[(size_t)m * 512 + tid];
    float v1 = (float)y[(size_t)m * 512 + 256 + tid];
    red[tid] = v0 + v1;
    __syncthreads();
    for (int s = 128; s > 0; s >>= 1) { if (tid < s) red[tid] += red[tid + s]; __syncthreads(); }
    float mean = red[0] * (1.f / 512.f);
    __syncthreads();
    float d0 = v0 - mean, d1 = v1 - mean;
    red[tid] = d0 * d0 + d1 * d1;
    __syncthreads();
    for (int s = 128; s > 0; s >>= 1) { if (tid < s) red[tid] += red[tid + s]; __syncthreads(); }
    float inv = rsqrtf(red[0] * (1.f / 512.f) + 1e-5f);
    out[(size_t)m * 512 + tid]       = (h16)(d0 * inv * g[tid] + bta[tid]);
    out[(size_t)m * 512 + 256 + tid] = (h16)(d1 * inv * g[256 + tid] + bta[256 + tid]);
}

// ---------------------------------------------------------------------------
// Second head layers: H1 [8192, 512] f32 (cols 0..255 binary, 256..511 multi).
// out: [0,8192) binary logits; [8192, 8192+8192*15) multi logits.
// ---------------------------------------------------------------------------
__global__ void k_heads(const float* __restrict__ H1,
                        const float* __restrict__ bh_b1, const float* __restrict__ bh_W2,
                        const float* __restrict__ bh_b2,
                        const float* __restrict__ mh_b1, const float* __restrict__ mh_W2,
                        const float* __restrict__ mh_b2, float* __restrict__ out) {
    int m = blockIdx.x * blockDim.x + threadIdx.x;
    if (m >= 8192) return;
    const float* r = H1 + (size_t)m * 512;
    float sb = bh_b2[0];
    for (int j = 0; j < 256; ++j) sb += fmaxf(r[j] + bh_b1[j], 0.f) * bh_W2[j];
    out[m] = sb;
    float acc[15];
#pragma unroll
    for (int l = 0; l < 15; ++l) acc[l] = mh_b2[l];
    for (int j = 0; j < 256; ++j) {
        float v = fmaxf(r[256 + j] + mh_b1[j], 0.f);
#pragma unroll
        for (int l = 0; l < 15; ++l) acc[l] += v * mh_W2[j * 15 + l];
    }
#pragma unroll
    for (int l = 0; l < 15; ++l) out[8192 + m * 15 + l] = acc[l];
}

// ---------------------------------------------------------------------------
extern "C" void kernel_launch(void* const* d_in, const int* in_sizes, int n_in,
                              void* d_out, int out_size, void* d_ws, size_t ws_size,
                              hipStream_t stream) {
    const int*   char_ids  = (const int*)  d_in[0];
    const int*   char2word = (const int*)  d_in[1];
    const int*   word_ids  = (const int*)  d_in[2];
    const float* feats     = (const float*)d_in[3];
    const float* tok       = (const float*)d_in[4];
    const float* emb_tab   = (const float*)d_in[5];
    const float* fp_W1 = (const float*)d_in[6],  *fp_b1 = (const float*)d_in[7];
    const float* fp_ln_g = (const float*)d_in[8], *fp_ln_b = (const float*)d_in[9];
    const float* fp_W2 = (const float*)d_in[10], *fp_b2 = (const float*)d_in[11];
    const float* lstm_Wih0  = (const float*)d_in[12];
    const float* lstm_Whh0  = (const float*)d_in[13];
    const float* lstm_b0    = (const float*)d_in[14];
    const float* lstm_Wih12 = (const float*)d_in[15];
    const float* lstm_Whh12 = (const float*)d_in[16];
    const float* lstm_b12   = (const float*)d_in[17];
    const float* ln_g = (const float*)d_in[18], *ln_b = (const float*)d_in[19];
    const float* bh_W1 = (const float*)d_in[20], *bh_b1 = (const float*)d_in[21];
    const float* bh_W2 = (const float*)d_in[22], *bh_b2 = (const float*)d_in[23];
    const float* mh_W1 = (const float*)d_in[24], *mh_b1 = (const float*)d_in[25];
    const float* mh_W2 = (const float*)d_in[26], *mh_b2 = (const float*)d_in[27];
    float* out = (float*)d_out;

    // workspace carve-up (256B aligned)
    char* ws = (char*)d_ws;
    size_t o = 0;
    auto take = [&](size_t bytes) { char* p = ws + o; o = (o + bytes + 255) & ~(size_t)255; return p; };
    float* Zpre    = (float*)take((size_t)8192 * 2048 * 4);
    float* wsums   = (float*)take((size_t)16 * 64 * 768 * 4);
    float* wcnts   = (float*)take((size_t)16 * 64 * 4);
    h16*   Xf      = (h16*)  take((size_t)8192 * 1120 * 2);
    h16*   Wih0f   = (h16*)  take((size_t)2048 * 1120 * 2);
    h16*   Whh0f   = (h16*)  take((size_t)2048 * 256 * 2);
    h16*   Wih12f  = (h16*)  take((size_t)2 * 2048 * 512 * 2);
    h16*   Whh12f  = (h16*)  take((size_t)2 * 2048 * 256 * 2);
    h16*   headW   = (h16*)  take((size_t)512 * 512 * 2);
    h16*   y1      = (h16*)  take((size_t)8192 * 512 * 2);
    h16*   y2      = (h16*)  take((size_t)8192 * 512 * 2);
    h16*   y3      = (h16*)  take((size_t)8192 * 512 * 2);
    h16*   yln     = (h16*)  take((size_t)8192 * 512 * 2);
    float* H1      = (float*)take((size_t)8192 * 512 * 4);

    auto cdiv = [](int a, int b) { return (a + b - 1) / b; };

    // 1) pooling sums/counts (zero then atomic accumulate)
    k_zero<<<cdiv(16 * 64 * 768, 256), 256, 0, stream>>>(wsums, 16 * 64 * 768);
    k_zero<<<cdiv(16 * 64, 256), 256, 0, stream>>>(wcnts, 16 * 64);
    k_pool<<<16 * 384, 256, 0, stream>>>(word_ids, tok, wsums, wcnts);

    // 2) weight conversions to f16
    k_cvt_pad<<<cdiv(2048 * 1120, 256), 256, 0, stream>>>(lstm_Wih0, Wih0f, 2048, 1116, 1120);
    k_cvt<<<cdiv(2048 * 256, 256), 256, 0, stream>>>(lstm_Whh0, Whh0f, 2048 * 256);
    k_cvt<<<cdiv(2 * 2048 * 512, 256), 256, 0, stream>>>(lstm_Wih12, Wih12f, 2 * 2048 * 512);
    k_cvt<<<cdiv(2 * 2048 * 256, 256), 256, 0, stream>>>(lstm_Whh12, Whh12f, 2 * 2048 * 256);
    k_cvt_T<<<cdiv(512 * 256, 256), 256, 0, stream>>>(bh_W1, headW, 512, 256, 0, 512);
    k_cvt_T<<<cdiv(512 * 256, 256), 256, 0, stream>>>(mh_W1, headW, 512, 256, 256, 512);

    // 3) build X (f16, K padded to 1120)
    k_prep<<<8192, 128, 0, stream>>>(char_ids, char2word, feats, emb_tab,
                                     fp_W1, fp_b1, fp_ln_g, fp_ln_b, fp_W2, fp_b2,
                                     wsums, wcnts, Xf);

    const size_t recur_smem = 16 * 1024 * 4 + 16 * 256 * 2; // 72KB

    // 4) layer 0: Zpre = X @ Wih0^T ; recurrence -> y1
    k_gemm_f16<<<dim3(2048 / 256, 8192 / 16), 128, 0, stream>>>(Xf, Wih0f, Zpre, 2048, 1120, 1120, 1120);
    k_lstm_recur<<<2, 1024, recur_smem, stream>>>(Zpre, Whh0f, lstm_b0, y1);

    // 5) layer 1
    k_gemm_f16<<<dim3(2048 / 256, 8192 / 16), 128, 0, stream>>>(y1, Wih12f, Zpre, 2048, 512, 512, 512);
    k_lstm_recur<<<2, 1024, recur_smem, stream>>>(Zpre, Whh12f, lstm_b12, y2);

    // 6) layer 2
    k_gemm_f16<<<dim3(2048 / 256, 8192 / 16), 128, 0, stream>>>(y2, Wih12f + (size_t)2048 * 512, Zpre, 2048, 512, 512, 512);
    k_lstm_recur<<<2, 1024, recur_smem, stream>>>(Zpre, Whh12f + (size_t)2048 * 256, lstm_b12 + 4096, y3);

    // 7) layer norm + head layer 1 (WMMA) + head layer 2 (VALU)
    k_layernorm512<<<8192, 256, 0, stream>>>(y3, ln_g, ln_b, yln);
    k_gemm_f16<<<dim3(512 / 256, 8192 / 16), 128, 0, stream>>>(yln, headW, H1, 512, 512, 512, 512);
    k_heads<<<8192 / 256, 256, 0, stream>>>(H1, bh_b1, bh_W2, bh_b2, mh_b1, mh_W2, mh_b2, out);
}
